// ParallelGNNModel_64020782514365
// MI455X (gfx1250) — matrix operand order repaired and checked by reading, not compiled, not run on hardware
//
#include <hip/hip_runtime.h>

// ---------------------------------------------------------------------------
// ParallelGNNModel forward for MI455X (gfx1250, wave32).
//  - All dense linears go through v_wmma_f32_16x16x32_bf16 (memory-bound GEMMs,
//    fp32 accumulate, bf16 operands). A-tile staged to LDS with
//    global_load_async_to_lds_b128 (ASYNCcnt path); K-chain fully unrolled.
//  - Edge passes are wave-per-edge scatter/gather (the real bottleneck:
//    ~2-3 GB irregular traffic per pass at 23.3 TB/s HBM).
// ---------------------------------------------------------------------------

#define HDIM     128
#define NDIMV    128
#define EDIM     32
#define GDIMV    64
#define NNODES   50000
#define NEDGES   1600000
#define NBATCH   500
#define EPSV     1e-5f

typedef __attribute__((ext_vector_type(16))) __bf16 v16bf;
typedef __attribute__((ext_vector_type(8)))  __bf16 v8bf;
typedef __attribute__((ext_vector_type(8)))  float  v8f;

__device__ __forceinline__ __bf16 f2bf(float f) { return (__bf16)f; }  // native v_cvt, RNE

// order-preserving float<->uint encoding for atomicMax on floats
__device__ __forceinline__ unsigned fenc(float f) {
  unsigned u = __builtin_bit_cast(unsigned, f);
  return u ^ ((u >> 31) ? 0xFFFFFFFFu : 0x80000000u);
}
__device__ __forceinline__ float fdec(unsigned e) {
  unsigned u = e ^ ((e >> 31) ? 0x80000000u : 0xFFFFFFFFu);
  return __builtin_bit_cast(float, u);
}
__device__ __forceinline__ float lrelu(float x) { return x > 0.f ? x : 0.2f * x; }

__device__ __forceinline__ float wave_sum(float v) {
  #pragma unroll
  for (int o = 16; o; o >>= 1) v += __shfl_xor(v, o, 32);
  return v;
}

// async 16B global->LDS copy (per-lane); tracked by ASYNCcnt
__device__ __forceinline__ void async_copy16(unsigned lds_byte_off, const float* gsrc) {
  asm volatile("global_load_async_to_lds_b128 %0, %1, off"
               :: "v"(lds_byte_off), "v"(gsrc) : "memory");
}
__device__ __forceinline__ void wait_async0() {
  asm volatile("s_wait_asynccnt 0" ::: "memory");
}

// ---------------------------------------------------------------------------
// fills
// ---------------------------------------------------------------------------
__global__ void fill_f32(float* p, float v, long n) {
  long i = blockIdx.x * (long)blockDim.x + threadIdx.x;
  if (i < n) p[i] = v;
}
__global__ void fill_u32(unsigned* p, unsigned v, long n) {
  long i = blockIdx.x * (long)blockDim.x + threadIdx.x;
  if (i < n) p[i] = v;
}

// ---------------------------------------------------------------------------
// Y[m][n] = sum_k X[m][k] * W[n][k] (+ bias[n])      -- i.e. X @ W.T + b
// K template in {64,128,192,256}; NOUT multiple of 16. bf16 WMMA, fp32 accum.
// One block = 16 rows; 8 waves cover the N tiles. A tile staged to LDS (f32)
// via async global->LDS copies on full tiles; bf16 convert at fragment build.
// ---------------------------------------------------------------------------
template <int K>
__global__ __launch_bounds__(256) void gemm_xwt(
    const float* __restrict__ X, const float* __restrict__ W,
    const float* __restrict__ bias, float* __restrict__ Y,
    int nrows, int NOUT)
{
  __shared__ __align__(16) float As[16 * K];
  const int tid = threadIdx.x;
  const int M0  = blockIdx.x * 16;
  const bool fullM = (M0 + 16 <= nrows);

  constexpr int KD4 = K / 4;              // float4 chunks per row
  if (fullM) {
    // async raw copy: 16 rows x K floats, 16B per lane per issue (ASYNCcnt)
    #pragma unroll
    for (int i = tid; i < 4 * K; i += 256) {
      int r = i / KD4, k4 = i - r * KD4;
      unsigned lds = (unsigned)(unsigned long long)&As[r * K + k4 * 4];
      async_copy16(lds, X + (long)(M0 + r) * K + k4 * 4);
    }
    wait_async0();
  } else {
    for (int i = tid; i < 4 * K; i += 256) {
      int r = i / KD4, k4 = i - r * KD4;
      int row = M0 + r;
      float4 v = (row < nrows) ? *(const float4*)(X + (long)row * K + k4 * 4)
                               : float4{0.f, 0.f, 0.f, 0.f};
      *(float4*)(&As[r * K + k4 * 4]) = v;
    }
  }
  __syncthreads();

  const int wave = tid >> 5;
  const int lane = tid & 31;
  const int hs   = lane >> 4;     // half-wave select
  const int rl   = lane & 15;     // row-local (A) / col-local (B)

  const int ntiles = NOUT >> 4;
  for (int tn = wave; tn < ntiles; tn += 8) {
    const int n = tn * 16 + rl;
    const float* wrow = W + (long)n * K;
    v8f acc = {};
    #pragma unroll
    for (int kk = 0; kk < K; kk += 32) {
      // A fragment: lane (rl,hs): elems 0..7 -> K=kk+hs*8+j ; 8..15 -> K=kk+16+hs*8+j
      const float* ar = &As[rl * K + kk + hs * 8];
      v16bf a;
      #pragma unroll
      for (int j = 0; j < 8; ++j) { a[j] = f2bf(ar[j]); a[j + 8] = f2bf(ar[16 + j]); }
      // B fragment: lane holds column n; elems 0..7 -> K=kk+hs*8+j ; 8..15 -> K=kk+16+hs*8+j
      const float* wr = wrow + kk + hs * 8;
      v16bf b;
      #pragma unroll
      for (int j = 0; j < 8; ++j) { b[j] = f2bf(wr[j]); b[j + 8] = f2bf(wr[16 + j]); }
      acc = __builtin_amdgcn_wmma_f32_16x16x32_bf16(
          false, a, false, b, (short)0, acc, false, false);
    }
    float bv = bias ? bias[n] : 0.f;
    if (fullM) {
      #pragma unroll
      for (int v = 0; v < 8; ++v) {        // C layout: row = M0 + v + 8*hs
        Y[(long)(M0 + v + 8 * hs) * NOUT + n] = acc[v] + bv;
      }
    } else {
      #pragma unroll
      for (int v = 0; v < 8; ++v) {
        int row = M0 + v + 8 * hs;
        if (row < nrows) Y[(long)row * NOUT + n] = acc[v] + bv;
      }
    }
  }
}

// ---------------------------------------------------------------------------
// per-row LayerNorm (+optional ReLU); wave per row; C in {64,128}
// ---------------------------------------------------------------------------
__global__ void ln_rows(const float* X, const float* __restrict__ g,
                        const float* __restrict__ b, float* Y,
                        int n, int C, int relu)
{
  int gid = blockIdx.x * blockDim.x + threadIdx.x;
  int row = gid >> 5, lane = gid & 31;
  if (row >= n) return;
  int per = C >> 5;
  float v[8];
  float s = 0.f;
  for (int j = 0; j < per; ++j) { v[j] = X[(long)row * C + lane + 32 * j]; s += v[j]; }
  s = wave_sum(s);
  float mean = s / C;
  float q = 0.f;
  for (int j = 0; j < per; ++j) { float d = v[j] - mean; q += d * d; }
  q = wave_sum(q);
  float istd = rsqrtf(q / C + EPSV);
  for (int j = 0; j < per; ++j) {
    int c = lane + 32 * j;
    float y = g[c] * (v[j] - mean) * istd + b[c];
    if (relu) y = fmaxf(y, 0.f);
    Y[(long)row * C + c] = y;
  }
}

// ---------------------------------------------------------------------------
// BatchNorm batch stats over rows. Requires gridDim*blockDim % C == 0
// (launched <<<512,256>>>, C in {64,128,256}).
// ---------------------------------------------------------------------------
__global__ void col_acc(const float* __restrict__ X, float* sums, float* sq,
                        long total, int C)
{
  long tid = blockIdx.x * (long)blockDim.x + threadIdx.x;
  long stride = (long)gridDim.x * blockDim.x;
  int c = (int)(tid % C);
  float s = 0.f, q = 0.f;
  for (long i = tid; i < total; i += stride) { float v = X[i]; s += v; q += v * v; }
  atomicAdd(&sums[c], s);
  atomicAdd(&sq[c], q);
}
__global__ void col_fin(const float* sums, const float* sq, float* mean,
                        float* istd, int C, float invn)
{
  int c = blockIdx.x * blockDim.x + threadIdx.x;
  if (c < C) {
    float m = sums[c] * invn;
    mean[c] = m;
    istd[c] = rsqrtf(sq[c] * invn - m * m + EPSV);
  }
}

// y = relu(bn(x)) (+ res); may run in place (X==Y)
__global__ void bn_apply(const float* X, const float* mean, const float* istd,
                         const float* g, const float* b, const float* res,
                         float* Y, long total, int C)
{
  long i = blockIdx.x * (long)blockDim.x + threadIdx.x;
  if (i >= total) return;
  int c = (int)(i % C);
  float y = g[c] * (X[i] - mean[c]) * istd[c] + b[c];
  y = fmaxf(y, 0.f);
  if (res) y += res[i];
  Y[i] = y;
}

// ---------------------------------------------------------------------------
// graph structure kernels
// ---------------------------------------------------------------------------
__global__ void edge_count(const int* __restrict__ tgt, float* deg, int E) {
  int e = blockIdx.x * blockDim.x + threadIdx.x;
  if (e < E) atomicAdd(&deg[tgt[e]], 1.f);
}
__global__ void node_prep(const float* deg, float* dis, float* selfc, int n) {
  int i = blockIdx.x * blockDim.x + threadIdx.x;
  if (i < n) { float d = rsqrtf(deg[i] + 1.f); dis[i] = d; selfc[i] = d * d; }
}

// conv[tgt] += xw[src] * dis[src]*dis[tgt]   (wave per edge, 4 floats/lane)
__global__ void gcn_scatter(const int* __restrict__ src, const int* __restrict__ tgt,
                            const float* __restrict__ dis,
                            const float* __restrict__ xw, float* conv, int E)
{
  long gid = blockIdx.x * (long)blockDim.x + threadIdx.x;
  int e = (int)(gid >> 5);
  if (e >= E) return;
  int lane = (int)(gid & 31);
  int s = src[e], t = tgt[e];
  float c = dis[s] * dis[t];
  const float* xs = xw + (long)s * HDIM;
  float* cv = conv + (long)t * HDIM;
  #pragma unroll
  for (int j = 0; j < 4; ++j) {
    int d = lane + 32 * j;
    atomicAdd(&cv[d], xs[d] * c);
  }
}

__global__ void gcn_self_bias(const float* __restrict__ xw, const float* selfc,
                              const float* bias, float* conv, long total)
{
  long i = blockIdx.x * (long)blockDim.x + threadIdx.x;
  if (i >= total) return;
  int r = (int)(i >> 7), c = (int)(i & 127);
  conv[i] += xw[i] * selfc[r] + bias[c];
}

// ---------------------------------------------------------------------------
// GATv2 edge pass 1: fused ea = edge_attr @ lin_e_w.T, loop_ea scatter,
// e = att . leaky_relu(xl[src]+xr[tgt]+ea), segment_max(e) via atomicMax
// ---------------------------------------------------------------------------
__global__ void gat_edge1(const int* __restrict__ src, const int* __restrict__ tgt,
                          const float* __restrict__ eattr, const float* __restrict__ We,
                          const float* __restrict__ att,
                          const float* __restrict__ xl, const float* __restrict__ xr,
                          float* loop_ea, float* e_edge, unsigned* menc, int E)
{
  long gid = blockIdx.x * (long)blockDim.x + threadIdx.x;
  int e = (int)(gid >> 5);
  if (e >= E) return;
  int lane = (int)(gid & 31);
  int s = src[e], t = tgt[e];
  const float4* ap = (const float4*)(eattr + (long)e * EDIM);
  float4 av[8];
  #pragma unroll
  for (int k = 0; k < 8; ++k) av[k] = ap[k];
  const float* xls = xl + (long)s * HDIM;
  const float* xrt = xr + (long)t * HDIM;
  float* lea = loop_ea + (long)t * HDIM;
  float part = 0.f;
  #pragma unroll
  for (int j = 0; j < 4; ++j) {
    int d = lane + 32 * j;
    const float4* wp = (const float4*)(We + (long)d * EDIM);
    float ea = 0.f;
    #pragma unroll
    for (int k = 0; k < 8; ++k) {
      float4 w = wp[k];
      ea += av[k].x * w.x + av[k].y * w.y + av[k].z * w.z + av[k].w * w.w;
    }
    atomicAdd(&lea[d], ea);
    part += lrelu(xls[d] + xrt[d] + ea) * att[d];
  }
  part = wave_sum(part);
  if (lane == 0) {
    e_edge[e] = part;
    atomicMax(&menc[t], fenc(part));
  }
}

// per-node: e_self, m = max(seg_max, e_self), z = aself, accum = xl*aself
__global__ void gat_node_mid(const float* deg, const float* __restrict__ loop_ea,
                             const float* __restrict__ xl, const float* __restrict__ xr,
                             const float* __restrict__ att, const unsigned* menc,
                             float* m_f, float* z, float* accum, int n)
{
  int gid = blockIdx.x * blockDim.x + threadIdx.x;
  int i = gid >> 5, lane = gid & 31;
  if (i >= n) return;
  float inv = 1.f / fmaxf(deg[i], 1.f);
  const float* xli = xl + (long)i * HDIM;
  const float* xri = xr + (long)i * HDIM;
  const float* lei = loop_ea + (long)i * HDIM;
  float part = 0.f;
  #pragma unroll
  for (int j = 0; j < 4; ++j) {
    int d = lane + 32 * j;
    part += lrelu(xli[d] + xri[d] + lei[d] * inv) * att[d];
  }
  part = wave_sum(part);
  float m = fmaxf(fdec(menc[i]), part);
  float aself = __expf(part - m);
  if (lane == 0) { m_f[i] = m; z[i] = aself; }
  float* ai = accum + (long)i * HDIM;
  #pragma unroll
  for (int j = 0; j < 4; ++j) { int d = lane + 32 * j; ai[d] = xli[d] * aself; }
}

// edge pass 2: ae = exp(e - m[tgt]); z[tgt]+=ae; accum[tgt]+=xl[src]*ae
__global__ void gat_edge2(const int* __restrict__ src, const int* __restrict__ tgt,
                          const float* __restrict__ e_edge, const float* __restrict__ m_f,
                          const float* __restrict__ xl, float* z, float* accum, int E)
{
  long gid = blockIdx.x * (long)blockDim.x + threadIdx.x;
  int e = (int)(gid >> 5);
  if (e >= E) return;
  int lane = (int)(gid & 31);
  int s = src[e], t = tgt[e];
  float ae = __expf(e_edge[e] - m_f[t]);
  if (lane == 0) atomicAdd(&z[t], ae);
  const float* xls = xl + (long)s * HDIM;
  float* at = accum + (long)t * HDIM;
  #pragma unroll
  for (int j = 0; j < 4; ++j) {
    int d = lane + 32 * j;
    atomicAdd(&at[d], xls[d] * ae);
  }
}

__global__ void gat_fin(float* accum, const float* z, const float* bias, long total) {
  long i = blockIdx.x * (long)blockDim.x + threadIdx.x;
  if (i >= total) return;
  int r = (int)(i >> 7), c = (int)(i & 127);
  accum[i] = accum[i] / z[r] + bias[c];
}

// pooling: wave per node, two segment-sums
__global__ void pool2(const int* __restrict__ batch, const float* __restrict__ g,
                      const float* __restrict__ a, float* pg, float* pa, int n)
{
  int gid = blockIdx.x * blockDim.x + threadIdx.x;
  int i = gid >> 5, lane = gid & 31;
  if (i >= n) return;
  int b = batch[i];
  const float* gi = g + (long)i * HDIM;
  const float* ai = a + (long)i * HDIM;
  float* pgb = pg + (long)b * HDIM;
  float* pab = pa + (long)b * HDIM;
  #pragma unroll
  for (int j = 0; j < 4; ++j) {
    int d = lane + 32 * j;
    atomicAdd(&pgb[d], gi[d]);
    atomicAdd(&pab[d], ai[d]);
  }
}

__global__ void concat_cols(const float* A, int Ca, const float* B, int Cb,
                            float* Y, int n)
{
  int C = Ca + Cb;
  long i = blockIdx.x * (long)blockDim.x + threadIdx.x;
  if (i >= (long)n * C) return;
  int r = (int)(i / C), c = (int)(i % C);
  Y[i] = (c < Ca) ? A[(long)r * Ca + c] : B[(long)r * Cb + (c - Ca)];
}

__global__ void final_dot(const float* __restrict__ X, const float* __restrict__ w,
                          const float* b, float* out, int n)
{
  int gid = blockIdx.x * blockDim.x + threadIdx.x;
  int r = gid >> 5, lane = gid & 31;
  if (r >= n) return;
  float s = X[(long)r * 64 + lane] * w[lane] + X[(long)r * 64 + lane + 32] * w[lane + 32];
  s = wave_sum(s);
  if (lane == 0) out[r] = s + b[0];
}

// ---------------------------------------------------------------------------
// host
// ---------------------------------------------------------------------------
extern "C" void kernel_launch(void* const* d_in, const int* in_sizes, int n_in,
                              void* d_out, int out_size, void* d_ws, size_t ws_size,
                              hipStream_t stream)
{
  (void)in_sizes; (void)n_in; (void)out_size; (void)ws_size;

  const float* x     = (const float*)d_in[0];
  const int*   ei    = (const int*)d_in[1];
  const int*   src   = ei;
  const int*   tgt   = ei + NEDGES;
  const float* eattr = (const float*)d_in[2];
  const int*   batch = (const int*)d_in[3];
  const float* gf    = (const float*)d_in[4];
  const float* P[64];
  for (int i = 5; i < 64; ++i) P[i] = (const float*)d_in[i];
  // 5 embed_w 6 embed_b 7 embed_ln_g 8 embed_ln_b
  // 9+6i gcn_w, 10+6i gcn_b, 11+6i bn_g, 12+6i bn_b, 13+6i res_w, 14+6i res_b
  // 27 lin_l_w 28 lin_l_b 29 lin_r_w 30 lin_r_b 31 lin_e_w 32 att 33 gat_bias
  // 34 gat_bn_g 35 gat_bn_b 36 res_gat_w 37 res_gat_b
  // 38..45 gm1/gm2, 46..53 fu1/fu2, 54..61 fc1/fc2, 62 fc3_w 63 fc3_b
  float* out = (float*)d_out;

  const long NH  = (long)NNODES * HDIM;
  const size_t NHB = (size_t)NH * 4;

  char* wsb = (char*)d_ws;
  size_t cur = 0;
  auto take = [&](size_t bytes) -> void* {
    void* p = wsb + cur;
    cur += (bytes + 255) & ~(size_t)255;
    return p;
  };
  float* h    = (float*)take(NHB);
  float* g    = (float*)take(NHB);
  float* t0   = (float*)take(NHB);     // xw / loop_ea / final 'a'
  float* t1   = (float*)take(NHB);     // conv / gat accum
  float* t2   = (float*)take(NHB);     // residual GEMM output
  float* xl   = (float*)take(NHB);
  float* xr   = (float*)take(NHB);
  float* deg  = (float*)take(NNODES * 4);
  float* dis  = (float*)take(NNODES * 4);
  float* selfc= (float*)take(NNODES * 4);
  float* m_f  = (float*)take(NNODES * 4);
  float* z    = (float*)take(NNODES * 4);
  unsigned* menc = (unsigned*)take(NNODES * 4);
  float* e_edge  = (float*)take((size_t)NEDGES * 4);
  float* sums = (float*)take(512 * 4);       // sums[256] + sumsq[256]
  float* sq   = sums + 256;
  float* meanb= (float*)take(256 * 4);
  float* istdb= (float*)take(256 * 4);
  float* pg   = (float*)take((size_t)NBATCH * HDIM * 4);
  float* pa   = (float*)take((size_t)NBATCH * HDIM * 4);
  float* fused= (float*)take((size_t)NBATCH * 256 * 4);
  float* fb1  = (float*)take((size_t)NBATCH * 256 * 4);
  float* fb2  = (float*)take((size_t)NBATCH * 128 * 4);
  float* gl1  = (float*)take((size_t)NBATCH * 128 * 4);
  float* gl2  = (float*)take((size_t)NBATCH * 64 * 4);
  float* feat = (float*)take((size_t)NBATCH * 192 * 4);
  float* f1   = (float*)take((size_t)NBATCH * 128 * 4);
  float* f2   = (float*)take((size_t)NBATCH * 64 * 4);

  auto cdiv = [](long a, long b) { return (int)((a + b - 1) / b); };
  const int gN    = cdiv(NNODES, 256);
  const int gNH   = cdiv(NH, 256);
  const int gE32  = cdiv((long)NEDGES * 32, 256);
  const int gN32  = cdiv((long)NNODES * 32, 256);
  const int gMt   = cdiv(NNODES, 16);
  const int gMtB  = cdiv(NBATCH, 16);

  // ---- degrees / norm coefficients ----
  fill_f32<<<gN, 256, 0, stream>>>(deg, 0.f, NNODES);
  edge_count<<<cdiv(NEDGES, 256), 256, 0, stream>>>(tgt, deg, NEDGES);
  node_prep<<<gN, 256, 0, stream>>>(deg, dis, selfc, NNODES);

  // ---- embed: h = relu(LN(x @ embed_w.T + b)) ----
  gemm_xwt<128><<<gMt, 256, 0, stream>>>(x, P[5], P[6], t0, NNODES, HDIM);
  ln_rows<<<gN32, 256, 0, stream>>>(t0, P[7], P[8], h, NNODES, HDIM, 1);

  // ---- GCN branch (3 layers) ----
  for (int i = 0; i < 3; ++i) {
    const float* Ain = (i == 0) ? h : g;
    gemm_xwt<128><<<gMt, 256, 0, stream>>>(Ain, P[9 + 6 * i], nullptr, t0, NNODES, HDIM);
    fill_f32<<<gNH, 256, 0, stream>>>(t1, 0.f, NH);
    gcn_scatter<<<gE32, 256, 0, stream>>>(src, tgt, dis, t0, t1, NEDGES);
    gcn_self_bias<<<gNH, 256, 0, stream>>>(t0, selfc, P[10 + 6 * i], t1, NH);
    fill_f32<<<2, 256, 0, stream>>>(sums, 0.f, 512);
    col_acc<<<512, 256, 0, stream>>>(t1, sums, sq, NH, HDIM);
    col_fin<<<1, 256, 0, stream>>>(sums, sq, meanb, istdb, HDIM, 1.f / NNODES);
    gemm_xwt<128><<<gMt, 256, 0, stream>>>(Ain, P[13 + 6 * i], P[14 + 6 * i], t2, NNODES, HDIM);
    bn_apply<<<gNH, 256, 0, stream>>>(t1, meanb, istdb, P[11 + 6 * i], P[12 + 6 * i], t2, g, NH, HDIM);
  }

  // ---- GATv2 branch ----
  gemm_xwt<128><<<gMt, 256, 0, stream>>>(h, P[27], P[28], xl, NNODES, HDIM);
  gemm_xwt<128><<<gMt, 256, 0, stream>>>(h, P[29], P[30], xr, NNODES, HDIM);
  fill_f32<<<gNH, 256, 0, stream>>>(t0, 0.f, NH);                 // loop_ea sum
  fill_u32<<<gN, 256, 0, stream>>>(menc, 0x007FFFFFu, NNODES);    // fenc(-inf)
  gat_edge1<<<gE32, 256, 0, stream>>>(src, tgt, eattr, P[31], P[32], xl, xr,
                                      t0, e_edge, menc, NEDGES);
  gat_node_mid<<<gN32, 256, 0, stream>>>(deg, t0, xl, xr, P[32], menc,
                                         m_f, z, t1, NNODES);
  gat_edge2<<<gE32, 256, 0, stream>>>(src, tgt, e_edge, m_f, xl, z, t1, NEDGES);
  gat_fin<<<gNH, 256, 0, stream>>>(t1, z, P[33], NH);
  fill_f32<<<2, 256, 0, stream>>>(sums, 0.f, 512);
  col_acc<<<512, 256, 0, stream>>>(t1, sums, sq, NH, HDIM);
  col_fin<<<1, 256, 0, stream>>>(sums, sq, meanb, istdb, HDIM, 1.f / NNODES);
  gemm_xwt<128><<<gMt, 256, 0, stream>>>(h, P[36], P[37], t2, NNODES, HDIM);
  bn_apply<<<gNH, 256, 0, stream>>>(t1, meanb, istdb, P[34], P[35], t2, t0, NH, HDIM); // a -> t0

  // ---- pooling ----
  fill_f32<<<cdiv((long)NBATCH * HDIM, 256), 256, 0, stream>>>(pg, 0.f, (long)NBATCH * HDIM);
  fill_f32<<<cdiv((long)NBATCH * HDIM, 256), 256, 0, stream>>>(pa, 0.f, (long)NBATCH * HDIM);
  pool2<<<gN32, 256, 0, stream>>>(batch, g, t0, pg, pa, NNODES);

  // ---- fusion MLP ----
  concat_cols<<<cdiv((long)NBATCH * 256, 256), 256, 0, stream>>>(pg, HDIM, pa, HDIM, fused, NBATCH);
  gemm_xwt<256><<<gMtB, 256, 0, stream>>>(fused, P[46], P[47], fb1, NBATCH, 256);
  fill_f32<<<2, 256, 0, stream>>>(sums, 0.f, 512);
  col_acc<<<512, 256, 0, stream>>>(fb1, sums, sq, (long)NBATCH * 256, 256);
  col_fin<<<1, 256, 0, stream>>>(sums, sq, meanb, istdb, 256, 1.f / NBATCH);
  bn_apply<<<cdiv((long)NBATCH * 256, 256), 256, 0, stream>>>(fb1, meanb, istdb, P[48], P[49],
                                                              nullptr, fb1, (long)NBATCH * 256, 256);
  gemm_xwt<256><<<gMtB, 256, 0, stream>>>(fb1, P[50], P[51], fb2, NBATCH, 128);
  fill_f32<<<2, 256, 0, stream>>>(sums, 0.f, 512);
  col_acc<<<512, 256, 0, stream>>>(fb2, sums, sq, (long)NBATCH * 128, 128);
  col_fin<<<1, 256, 0, stream>>>(sums, sq, meanb, istdb, 128, 1.f / NBATCH);
  bn_apply<<<cdiv((long)NBATCH * 128, 256), 256, 0, stream>>>(fb2, meanb, istdb, P[52], P[53],
                                                              nullptr, fb2, (long)NBATCH * 128, 128);

  // ---- global-features MLP ----
  gemm_xwt<64><<<gMtB, 256, 0, stream>>>(gf, P[38], P[39], gl1, NBATCH, 128);
  ln_rows<<<cdiv((long)NBATCH * 32, 256), 256, 0, stream>>>(gl1, P[40], P[41], gl1, NBATCH, 128, 1);
  gemm_xwt<128><<<gMtB, 256, 0, stream>>>(gl1, P[42], P[43], gl2, NBATCH, 64);
  ln_rows<<<cdiv((long)NBATCH * 32, 256), 256, 0, stream>>>(gl2, P[44], P[45], gl2, NBATCH, 64, 1);

  // ---- head ----
  concat_cols<<<cdiv((long)NBATCH * 192, 256), 256, 0, stream>>>(fb2, 128, gl2, 64, feat, NBATCH);
  gemm_xwt<192><<<gMtB, 256, 0, stream>>>(feat, P[54], P[55], f1, NBATCH, 128);
  fill_f32<<<2, 256, 0, stream>>>(sums, 0.f, 512);
  col_acc<<<512, 256, 0, stream>>>(f1, sums, sq, (long)NBATCH * 128, 128);
  col_fin<<<1, 256, 0, stream>>>(sums, sq, meanb, istdb, 128, 1.f / NBATCH);
  bn_apply<<<cdiv((long)NBATCH * 128, 256), 256, 0, stream>>>(f1, meanb, istdb, P[56], P[57],
                                                              nullptr, f1, (long)NBATCH * 128, 128);
  gemm_xwt<128><<<gMtB, 256, 0, stream>>>(f1, P[58], P[59], f2, NBATCH, 64);
  fill_f32<<<2, 256, 0, stream>>>(sums, 0.f, 512);
  col_acc<<<512, 256, 0, stream>>>(f2, sums, sq, (long)NBATCH * 64, 64);
  col_fin<<<1, 256, 0, stream>>>(sums, sq, meanb, istdb, 64, 1.f / NBATCH);
  bn_apply<<<cdiv((long)NBATCH * 64, 256), 256, 0, stream>>>(f2, meanb, istdb, P[60], P[61],
                                                             nullptr, f2, (long)NBATCH * 64, 64);
  final_dot<<<cdiv((long)NBATCH * 32, 256), 256, 0, stream>>>(f2, P[62], P[63], out, NBATCH);
}
